// WalzeAttentionLayerV3_12644383719814
// MI455X (gfx1250) — matrix operand order repaired and checked by reading, not compiled, not run on hardware
//
#include <hip/hip_runtime.h>
#include <hip/hip_bf16.h>

typedef __attribute__((ext_vector_type(16))) _Float16 v16h;
typedef __attribute__((ext_vector_type(8)))  _Float16 v8h;
typedef __attribute__((ext_vector_type(8)))  float    v8f;

#define NB    8       // batch
#define NC    512     // channels
#define NL    1024    // H*W
#define NHEAD 8
#define HD    64
#define VPAD  16
#define VSTRIDE 1072  // 16 pad + 1024 + 32 pad, multiple of 8 halves

__device__ __forceinline__ float gelu_f(float x) {
  return 0.5f * x * (1.0f + erff(x * 0.70710678118654752f));
}
__device__ __forceinline__ float sigmoid_f(float x) {
  return 1.0f / (1.0f + __expf(-x));
}

// A tile 16(M) x 32(K) f16 from row-major [row][k] memory.
// ISA layout: lane<16 -> M=lane, K in {0..7, 16..23}; lane>=16 -> M=lane-16, K in {8..15, 24..31}
__device__ __forceinline__ v16h load_a16x32(const _Float16* p, int ld, int lane) {
  const _Float16* q = p + (lane & 15) * ld + ((lane >> 4) << 3);
  v8h lo = *(const v8h*)(q);
  v8h hi = *(const v8h*)(q + 16);
  v16h r;
#pragma unroll
  for (int i = 0; i < 8; ++i) { r[i] = lo[i]; r[i + 8] = hi[i]; }
  return r;
}
// B tile 32(K) x 16(N) f16 from row-major BT [n][k] memory (column n of B contiguous in K).
// ISA layout: lane<16 -> N=lane, K in {0..7, 16..23}; lane>=16 -> N=lane-16, K in {8..15, 24..31}
__device__ __forceinline__ v16h load_b32x16(const _Float16* p, int ld, int lane) {
  const _Float16* q = p + (lane & 15) * ld + ((lane >> 4) << 3);
  v8h lo = *(const v8h*)(q);
  v8h hi = *(const v8h*)(q + 16);
  v16h r;
#pragma unroll
  for (int i = 0; i < 8; ++i) { r[i] = lo[i]; r[i + 8] = hi[i]; }
  return r;
}

__device__ __forceinline__ v8f wmma_f16(v16h a, v16h b, v8f c) {
  return __builtin_amdgcn_wmma_f32_16x16x32_f16(false, a, false, b, (short)0, c, false, false);
}

// 32(M) x 32(N) register-blocked macro-tile: 4 WMMAs per 4 tile loads (2A + 2B)
struct CTile4 { v8f c00, c01, c10, c11; };
__device__ __forceinline__ CTile4 gemm_32x32(const _Float16* A, int lda,
                                             const _Float16* Bt, int ldb,
                                             int K, int lane) {
  CTile4 r; r.c00 = {}; r.c01 = {}; r.c10 = {}; r.c11 = {};
  const _Float16* A1 = A + (size_t)16 * lda;
  const _Float16* B1 = Bt + (size_t)16 * ldb;
  for (int k = 0; k < K; k += 32) {
    v16h a0 = load_a16x32(A + k, lda, lane);
    v16h a1 = load_a16x32(A1 + k, lda, lane);
    v16h b0 = load_b32x16(Bt + k, ldb, lane);
    v16h b1 = load_b32x16(B1 + k, ldb, lane);
    r.c00 = wmma_f16(a0, b0, r.c00);
    r.c01 = wmma_f16(a0, b1, r.c01);
    r.c10 = wmma_f16(a1, b0, r.c10);
    r.c11 = wmma_f16(a1, b1, r.c11);
  }
  return r;
}

// ---------------- f32 -> f16 convert ----------------
__global__ void cvt_kernel(const float* __restrict__ src, _Float16* __restrict__ dst, int n) {
  int i = blockIdx.x * blockDim.x + threadIdx.x;
  if (i < n) dst[i] = (_Float16)src[i];
}

// ---------------- fused DualWalze: circular dw-conv (main+shared), mix, BN, GELU ----------------
// one block per (b,c); writes x_walze f32 (B,C,H,W) and f16 comb (B,L,2C) = [x | gelu(bn)]
__global__ void walze_kernel(const float* __restrict__ x,
                             const float* __restrict__ dww, const float* __restrict__ dwb,
                             const float* __restrict__ shw, const float* __restrict__ shb,
                             const float* __restrict__ mixw,
                             const float* __restrict__ gamma, const float* __restrict__ beta,
                             float* __restrict__ xw, _Float16* __restrict__ comb) {
  int bc = blockIdx.x;              // b*512 + c
  int b = bc >> 9, c = bc & 511;
  const float* xin = x + (size_t)bc * NL;
  float km[9], ks[9];
#pragma unroll
  for (int i = 0; i < 9; ++i) { km[i] = dww[c * 9 + i]; ks[i] = shw[i]; }
  float mix  = sigmoid_f(mixw[0]);
  float g    = gamma[c] * rsqrtf(1.0f + 1e-5f);
  float bt   = beta[c];
  float bm   = dwb[c], bs = shb[0];
  for (int l = threadIdx.x; l < NL; l += blockDim.x) {
    int h = l >> 5, w = l & 31;
    int hm = (h + 31) & 31, hp = (h + 1) & 31;
    int wl = (w + 31) & 31, wr = (w + 1) & 31;
    float n0 = xin[hm * 32 + wl], n1 = xin[hm * 32 + w], n2 = xin[hm * 32 + wr];
    float n3 = xin[h  * 32 + wl], n4 = xin[h  * 32 + w], n5 = xin[h  * 32 + wr];
    float n6 = xin[hp * 32 + wl], n7 = xin[hp * 32 + w], n8 = xin[hp * 32 + wr];
    float mo = bm + n0*km[0]+n1*km[1]+n2*km[2]+n3*km[3]+n4*km[4]+n5*km[5]+n6*km[6]+n7*km[7]+n8*km[8];
    float so = bs + n0*ks[0]+n1*ks[1]+n2*ks[2]+n3*ks[3]+n4*ks[4]+n5*ks[5]+n6*ks[6]+n7*ks[7]+n8*ks[8];
    float cb = mix * mo + (1.0f - mix) * so + n4;          // residual
    float gl = gelu_f(g * cb + bt);
    xw[(size_t)bc * NL + l] = gl;
    size_t base = ((size_t)(b * NL + l)) * (2 * NC);
    comb[base + c]      = (_Float16)n4;   // x_before
    comb[base + NC + c] = (_Float16)gl;   // x_after
  }
}

// ---------------- detector GEMM1: h = gelu(comb(8192x1024) * W1^T), WMMA 32x32 blocked ------
__global__ void det_gemm1_kernel(const _Float16* __restrict__ comb, const _Float16* __restrict__ w1h,
                                 const float* __restrict__ b1, float* __restrict__ hbuf) {
  int gid = blockIdx.x * blockDim.x + threadIdx.x;
  int wave = gid >> 5, lane = gid & 31;
  int nt2 = wave & 3, mt2 = wave >> 2;               // nt2<4 (N=128), mt2<256 (M=8192)
  const _Float16* A  = comb + (size_t)mt2 * 32 * (2 * NC);
  const _Float16* Bt = w1h + (size_t)nt2 * 32 * (2 * NC);
  CTile4 r = gemm_32x32(A, 2 * NC, Bt, 2 * NC, 2 * NC, lane);
  int nn = lane & 15, mb = (lane >> 4) * 8;
  const v8f* cs[4] = { &r.c00, &r.c01, &r.c10, &r.c11 };
#pragma unroll
  for (int t = 0; t < 4; ++t) {
    int ti = t >> 1, tj = t & 1;
    int col = nt2 * 32 + tj * 16 + nn;
    float bias = b1[col];
#pragma unroll
    for (int v = 0; v < 8; ++v) {
      int row = mt2 * 32 + ti * 16 + mb + v;
      hbuf[(size_t)row * 128 + col] = gelu_f((*cs[t])[v] + bias);
    }
  }
}

// ---------------- detector layer2 + sigmoid: one wave per row ----------------
__global__ void spike_kernel(const float* __restrict__ hbuf, const float* __restrict__ w2,
                             const float* __restrict__ b2, float* __restrict__ maskb) {
  int gid  = blockIdx.x * blockDim.x + threadIdx.x;
  int row  = gid >> 5, lane = gid & 31;              // 8192 rows
  float s = 0.0f;
#pragma unroll
  for (int j = 0; j < 4; ++j)
    s += hbuf[(size_t)row * 128 + lane + j * 32] * w2[lane + j * 32];
#pragma unroll
  for (int off = 16; off > 0; off >>= 1) s += __shfl_down(s, off, 32);
  if (lane == 0) maskb[row] = sigmoid_f(s + b2[0]);
}

// ---------------- QKV GEMM (8192 x 1536, K=512), 32x32 blocked, head-major scatter ----------
__global__ void qkv_gemm_kernel(const _Float16* __restrict__ comb, const _Float16* __restrict__ wqkv,
                                const float* __restrict__ bqkv,
                                _Float16* __restrict__ qh, _Float16* __restrict__ kh,
                                _Float16* __restrict__ vh) {
  int gid = blockIdx.x * blockDim.x + threadIdx.x;
  int wave = gid >> 5, lane = gid & 31;
  int nt2 = wave % 48, mt2 = wave / 48;              // nt2<48 (N=1536), mt2<256
  const _Float16* A  = comb + (size_t)mt2 * 32 * (2 * NC) + NC;  // x_after half, lda=1024
  const _Float16* Bt = wqkv + (size_t)nt2 * 32 * NC;
  CTile4 r = gemm_32x32(A, 2 * NC, Bt, NC, NC, lane);
  int nn = lane & 15, mb = (lane >> 4) * 8;
  const v8f* cs[4] = { &r.c00, &r.c01, &r.c10, &r.c11 };
#pragma unroll
  for (int t = 0; t < 4; ++t) {
    int ti = t >> 1, tj = t & 1;
    int f = nt2 * 32 + tj * 16 + nn;
    float bias = bqkv[f];
    int sec = f >> 9, fh = f & 511;
    int head = fh >> 6, d = fh & 63;
#pragma unroll
    for (int v = 0; v < 8; ++v) {
      int row = mt2 * 32 + ti * 16 + mb + v;         // b*1024 + l
      int b = row >> 10, l = row & 1023;
      _Float16 val = (_Float16)((*cs[t])[v] + bias);
      int bh = b * NHEAD + head;
      if (sec == 0)      qh[((size_t)bh * NL + l) * HD + d] = val;
      else if (sec == 1) kh[((size_t)bh * NL + l) * HD + d] = val;
      else               vh[((size_t)bh * HD + d) * VSTRIDE + VPAD + l] = val; // transposed+padded
    }
  }
}

// ---------------- banded attention: one wave per 16-query block ----------------
__global__ void attn_kernel(const _Float16* __restrict__ qh, const _Float16* __restrict__ kh,
                            const _Float16* __restrict__ vh, _Float16* __restrict__ ctx) {
  __shared__ __align__(16) float    sS[4][16][48];
  __shared__ __align__(16) _Float16 sP[4][16][64];
  int lane = threadIdx.x & 31, wv = threadIdx.x >> 5;
  int gb = blockIdx.x * 4 + wv;                      // 0..4095
  int m0 = (gb & 63) << 4;                           // query block start
  int bh = gb >> 6;                                  // b*8 + head
  int nn = lane & 15, mb = (lane >> 4) * 8;

  const _Float16* Q = qh + ((size_t)bh * NL + m0) * HD;
  v16h aq0 = load_a16x32(Q, HD, lane);
  v16h aq1 = load_a16x32(Q + 32, HD, lane);

  // scores over 3 key tiles: keys m0-16 .. m0+31 (band = +-16)
  for (int t = 0; t < 3; ++t) {
    int k0 = m0 + (t - 1) * 16;
    if (k0 < 0 || k0 >= NL) {
#pragma unroll
      for (int v = 0; v < 8; ++v) sS[wv][mb + v][t * 16 + nn] = -1e30f;
    } else {
      const _Float16* Kp = kh + ((size_t)bh * NL + k0) * HD;
      v8f c = {};
      c = wmma_f16(aq0, load_b32x16(Kp, HD, lane), c);
      c = wmma_f16(aq1, load_b32x16(Kp + 32, HD, lane), c);
#pragma unroll
      for (int v = 0; v < 8; ++v) sS[wv][mb + v][t * 16 + nn] = c[v] * 0.125f; // 1/sqrt(64)
    }
  }
  __syncthreads();

  // softmax over band [qq, qq+32] per row; write f16 probs, zero-padded to 64 cols
  if (lane < 16) {
    int qq = lane;
    float mx = -1e30f;
    for (int c2 = qq; c2 <= qq + 32; ++c2) mx = fmaxf(mx, sS[wv][qq][c2]);
#pragma unroll
    for (int c2 = 0; c2 < 64; ++c2) sP[wv][qq][c2] = (_Float16)0.0f;
    float sum = 0.0f;
    for (int c2 = qq; c2 <= qq + 32; ++c2) {
      float e = __expf(sS[wv][qq][c2] - mx);
      sS[wv][qq][c2] = e; sum += e;
    }
    float inv = 1.0f / sum;
    for (int c2 = qq; c2 <= qq + 32; ++c2)
      sP[wv][qq][c2] = (_Float16)(sS[wv][qq][c2] * inv);
  }
  __syncthreads();

  // ctx = P(16x64) x V(64x64); V stored (hd, L) so B-tiles are contiguous in K(=key)
  v16h pa0 = load_a16x32(&sP[wv][0][0], 64, lane);
  v16h pa1 = load_a16x32(&sP[wv][0][32], 64, lane);
  int b = bh >> 3, head = bh & 7;
  for (int dt = 0; dt < 4; ++dt) {
    const _Float16* Vb = vh + ((size_t)bh * HD + dt * 16) * VSTRIDE + VPAD + (m0 - 16);
    v8f c = {};
    c = wmma_f16(pa0, load_b32x16(Vb, VSTRIDE, lane), c);
    c = wmma_f16(pa1, load_b32x16(Vb + 32, VSTRIDE, lane), c);
#pragma unroll
    for (int v = 0; v < 8; ++v) {
      int l = m0 + mb + v;
      ctx[((size_t)b * NL + l) * NC + head * HD + dt * 16 + nn] = (_Float16)c[v];
    }
  }
}

// ---------------- out-proj GEMM, 32x32 blocked, fused gated residual combine -> d_out -------
__global__ void outproj_kernel(const _Float16* __restrict__ ctx, const _Float16* __restrict__ wout,
                               const float* __restrict__ bout, const float* __restrict__ xw,
                               const float* __restrict__ maskb, float* __restrict__ out) {
  int gid = blockIdx.x * blockDim.x + threadIdx.x;
  int wave = gid >> 5, lane = gid & 31;
  int nt2 = wave & 15, mt2 = wave >> 4;              // nt2<16 (N=512), mt2<256
  const _Float16* A  = ctx + (size_t)mt2 * 32 * NC;
  const _Float16* Bt = wout + (size_t)nt2 * 32 * NC;
  CTile4 r = gemm_32x32(A, NC, Bt, NC, NC, lane);
  int nn = lane & 15, mb = (lane >> 4) * 8;
  const v8f* cs[4] = { &r.c00, &r.c01, &r.c10, &r.c11 };
#pragma unroll
  for (int t = 0; t < 4; ++t) {
    int ti = t >> 1, tj = t & 1;
    int n = nt2 * 32 + tj * 16 + nn;
    float bias = bout[n];
#pragma unroll
    for (int v = 0; v < 8; ++v) {
      int row = mt2 * 32 + ti * 16 + mb + v;
      int b = row >> 10, l = row & 1023;
      size_t oi = ((size_t)(b * NC + n)) * NL + l;   // (B,C,H,W)
      out[oi] = xw[oi] + maskb[row] * ((*cs[t])[v] + bias);
    }
  }
}

extern "C" void kernel_launch(void* const* d_in, const int* in_sizes, int n_in,
                              void* d_out, int out_size, void* d_ws, size_t ws_size,
                              hipStream_t stream) {
  (void)in_sizes; (void)n_in; (void)out_size; (void)ws_size;
  const float* x     = (const float*)d_in[0];
  const float* dww   = (const float*)d_in[1];
  const float* dwb   = (const float*)d_in[2];
  const float* shw   = (const float*)d_in[3];
  const float* shb   = (const float*)d_in[4];
  const float* mixw  = (const float*)d_in[5];
  const float* gamma = (const float*)d_in[6];
  const float* beta  = (const float*)d_in[7];
  const float* dw1   = (const float*)d_in[8];
  const float* db1   = (const float*)d_in[9];
  const float* dw2   = (const float*)d_in[10];
  const float* db2   = (const float*)d_in[11];
  const float* wq    = (const float*)d_in[12];
  const float* bq    = (const float*)d_in[13];
  const float* wo    = (const float*)d_in[14];
  const float* bo    = (const float*)d_in[15];

  char* ws = (char*)d_ws;
  float*    xw    = (float*)ws;     ws += (size_t)NB * NC * NL * 4;            // 16.78 MB
  _Float16* comb  = (_Float16*)ws;  ws += (size_t)NB * NL * 2 * NC * 2;        // 16.78 MB
  float*    hbuf  = (float*)ws;     ws += (size_t)NB * NL * 128 * 4;           //  4.19 MB
  float*    maskb = (float*)ws;     ws += (size_t)NB * NL * 4;                 //  32 KB
  _Float16* qh    = (_Float16*)ws;  ws += (size_t)NB * NHEAD * NL * HD * 2;    //  8.39 MB
  _Float16* kh    = (_Float16*)ws;  ws += (size_t)NB * NHEAD * NL * HD * 2;    //  8.39 MB
  size_t vh_bytes = (size_t)NB * NHEAD * HD * VSTRIDE * 2;                     //  8.78 MB
  _Float16* vh    = (_Float16*)ws;  ws += vh_bytes;
  _Float16* ctxh  = (_Float16*)ws;  ws += (size_t)NB * NL * NC * 2;            //  8.39 MB
  _Float16* w1h   = (_Float16*)ws;  ws += (size_t)128 * 1024 * 2;
  _Float16* wqkvh = (_Float16*)ws;  ws += (size_t)1536 * 512 * 2;
  _Float16* wouth = (_Float16*)ws;  ws += (size_t)512 * 512 * 2;

  // zero V (covers the +-key padding the banded ctx matmul reads through)
  hipMemsetAsync(vh, 0, vh_bytes, stream);

  cvt_kernel<<<(128 * 1024 + 255) / 256, 256, 0, stream>>>(dw1, w1h, 128 * 1024);
  cvt_kernel<<<(1536 * 512 + 255) / 256, 256, 0, stream>>>(wq, wqkvh, 1536 * 512);
  cvt_kernel<<<(512 * 512 + 255) / 256, 256, 0, stream>>>(wo, wouth, 512 * 512);

  walze_kernel<<<NB * NC, 256, 0, stream>>>(x, dww, dwb, shw, shb, mixw, gamma, beta, xw, comb);

  det_gemm1_kernel<<<128, 256, 0, stream>>>(comb, w1h, db1, hbuf);   // 1024 waves
  spike_kernel<<<1024, 256, 0, stream>>>(hbuf, dw2, db2, maskb);     // 8192 waves

  qkv_gemm_kernel<<<1536, 256, 0, stream>>>(comb, wqkvh, bq, qh, kh, vh);  // 12288 waves
  attn_kernel<<<1024, 128, 0, stream>>>(qh, kh, vh, ctxh);                 // 4096 waves
  outproj_kernel<<<512, 256, 0, stream>>>(ctxh, wouth, bo, xw, maskb, (float*)d_out);
}
